// IBGCN_34978213659213
// MI455X (gfx1250) — matrix operand order repaired and checked by reading, not compiled, not run on hardware
//
#include <hip/hip_runtime.h>
#include <hip/hip_bf16.h>

// ---------------------------------------------------------------------------
// Types for CDNA5 WMMA (gfx1250, wave32)
// ---------------------------------------------------------------------------
typedef __bf16 bf16_t;
typedef __bf16 v16bf __attribute__((ext_vector_type(16)));
typedef float  v8f   __attribute__((ext_vector_type(8)));
typedef float  v4f   __attribute__((ext_vector_type(4)));

#define ACT_NONE 0
#define ACT_SIG  1
#define ACT_RELU 2
#define ACT_SP5  3   // softplus(x - 5)

template<int ACT>
__device__ __forceinline__ float act_apply(float v) {
    if (ACT == ACT_SIG) {
        // fast sigmoid: v_exp_f32 + v_rcp_f32 (avoid IEEE div expansion)
        return __builtin_amdgcn_rcpf(1.0f + __expf(-v));
    }
    if (ACT == ACT_RELU) return fmaxf(v, 0.0f);
    if (ACT == ACT_SP5) {
        float t = v - 5.0f;
        return (t > 20.0f) ? t : __logf(1.0f + __expf(t));
    }
    return v;
}

// Build a 16x32 bf16 A-fragment for one lane from row-major f32 activations.
// Layout (ISA 7.12.2, 16-bit A 16x32): lanes 0-15 -> M=lane, kb=0; lanes 16-31
// -> M=lane-16, kb=8. elems 0..7 -> K=kb+0..7 ; elems 8..15 -> K=kb+16..23.
__device__ __forceinline__ v16bf load_a_frag_fast(const float* __restrict__ p) {
    v4f x0 = *(const v4f*)(p + 0);
    v4f x1 = *(const v4f*)(p + 4);
    v4f x2 = *(const v4f*)(p + 16);
    v4f x3 = *(const v4f*)(p + 20);
    v16bf a;
#pragma unroll
    for (int i = 0; i < 4; ++i) {
        a[i]      = (bf16_t)x0[i];
        a[4 + i]  = (bf16_t)x1[i];
        a[8 + i]  = (bf16_t)x2[i];
        a[12 + i] = (bf16_t)x3[i];
    }
    return a;
}

__device__ __forceinline__ v16bf load_a_frag_pred(const float* __restrict__ p, bool valid) {
    v4f z = {0.f, 0.f, 0.f, 0.f};
    v4f x0 = valid ? *(const v4f*)(p + 0)  : z;
    v4f x1 = valid ? *(const v4f*)(p + 4)  : z;
    v4f x2 = valid ? *(const v4f*)(p + 16) : z;
    v4f x3 = valid ? *(const v4f*)(p + 20) : z;
    v16bf a;
#pragma unroll
    for (int i = 0; i < 4; ++i) {
        a[i]      = (bf16_t)x0[i];
        a[4 + i]  = (bf16_t)x1[i];
        a[8 + i]  = (bf16_t)x2[i];
        a[12 + i] = (bf16_t)x3[i];
    }
    return a;
}

// ---------------------------------------------------------------------------
// Generic WMMA GEMM: C[nrows, ncols] = act(A[nrows, KC*32] @ Wp + bias)
//  - A: row-major f32, leading dim ldA (>= KC*32)
//  - Wp: pre-packed bf16 B fragments, [kc][ct][lane] of v16bf
//  - 256 threads = 8 waves; block covers 128 rows; wave w owns rows +w*16.
//  - COLPAD: true only when ncols is not a multiple of 16 (final 16->10 layer)
// ---------------------------------------------------------------------------
template<int KC, int NT, int ACT, bool HAS_BIAS, bool COLPAD>
__global__ __launch_bounds__(256) void k_gemm(
    const float* __restrict__ A, int ldA,
    const v16bf* __restrict__ Wp,
    const float* __restrict__ bias,
    float* __restrict__ C, int ldC,
    int nrows, int ncols)
{
    const int lane    = threadIdx.x & 31;
    const int wave    = threadIdx.x >> 5;
    const int rowBase = blockIdx.x * 128 + wave * 16;

    const bool blockFull = ((int)(blockIdx.x + 1) * 128 <= nrows);  // wave-uniform

    // A-fragment row for this lane
    const int mrow = rowBase + (lane & 15);
    const int kb   = (lane & 16) ? 8 : 0;

    v16bf af[KC];
    if (blockFull) {
#pragma unroll
        for (int kc = 0; kc < KC; ++kc)
            af[kc] = load_a_frag_fast(A + (size_t)mrow * ldA + kc * 32 + kb);
    } else {
        const bool avalid = (mrow < nrows);
#pragma unroll
        for (int kc = 0; kc < KC; ++kc)
            af[kc] = load_a_frag_pred(A + (size_t)mrow * ldA + kc * 32 + kb, avalid);
    }

    const int colLane = lane & 15;
    const int rbase   = rowBase + ((lane & 16) ? 8 : 0);
    float* __restrict__ Crow = C + (size_t)rbase * ldC + colLane;

#pragma unroll
    for (int ct = 0; ct < NT; ++ct) {
        v8f acc = {};
#pragma unroll
        for (int kc = 0; kc < KC; ++kc) {
            v16bf b = Wp[(size_t)(kc * NT + ct) * 32 + lane];
            acc = __builtin_amdgcn_wmma_f32_16x16x32_bf16(
                false, af[kc], false, b, (short)0, acc, false, false);
        }
        const int  col   = ct * 16 + colLane;
        const bool colOk = (!COLPAD) || (col < ncols);
        float bv = 0.0f;
        if (HAS_BIAS && colOk) bv = bias[col];

        if (blockFull) {
            if (colOk) {
                float* Cp = Crow + ct * 16;
#pragma unroll
                for (int r = 0; r < 8; ++r) {
                    *Cp = act_apply<ACT>(acc[r] + bv);
                    Cp += ldC;
                }
            }
        } else {
#pragma unroll
            for (int r = 0; r < 8; ++r) {
                const int m = rbase + r;
                if (m < nrows && colOk)
                    C[(size_t)m * ldC + col] = act_apply<ACT>(acc[r] + bv);
            }
        }
    }
}

// ---------------------------------------------------------------------------
// Pack row-major f32 weights [Kreal, Nreal] into bf16 B fragments.
// B 32x16 layout: lane&15 = column; lanes 0-15 K=0..7,16..23, lanes 16-31 K=8..15,24..31.
// Out-of-range (K or N padding) entries are exactly bf16 zero.
// ---------------------------------------------------------------------------
__global__ void k_pack(const float* __restrict__ W, v16bf* __restrict__ Wp,
                       int Kreal, int Nreal, int KC, int NT)
{
    int idx   = blockIdx.x * blockDim.x + threadIdx.x;
    int total = KC * NT * 32;
    if (idx >= total) return;
    int lane = idx & 31;
    int frag = idx >> 5;
    int ct   = frag % NT;
    int kc   = frag / NT;
    int col  = ct * 16 + (lane & 15);
    int kb   = kc * 32 + ((lane & 16) ? 8 : 0);
    v16bf v;
#pragma unroll
    for (int t = 0; t < 16; ++t) {
        int kk = kb + (t < 8 ? t : t + 8);
        float x = (kk < Kreal && col < Nreal) ? W[(size_t)kk * Nreal + col] : 0.0f;
        v[t] = (bf16_t)x;
    }
    Wp[idx] = v;
}

// ---------------------------------------------------------------------------
// Graph norm / aggregation / misc elementwise kernels
// ---------------------------------------------------------------------------
__global__ void k_degree(const int* __restrict__ ei, float* __restrict__ deg, int E) {
    int e = blockIdx.x * blockDim.x + threadIdx.x;
    if (e >= E) return;
    atomicAdd(&deg[ei[(size_t)E + e]], 1.0f);
}

__global__ void k_dinv(const float* __restrict__ deg, float* __restrict__ dinv,
                       const int* __restrict__ batch, float* __restrict__ cnt, int N) {
    int n = blockIdx.x * blockDim.x + threadIdx.x;
    if (n >= N) return;
    dinv[n] = rsqrtf(deg[n] + 1.0f);           // +1 self-loop
    atomicAdd(&cnt[batch[n]], 1.0f);
}

// agg = g * snorm + b_gcn  (snorm = dinv^2)
__global__ void k_agg_init(const float* __restrict__ g, const float* __restrict__ dinv,
                           const float* __restrict__ b_gcn, float* __restrict__ agg, int N) {
    int id = blockIdx.x * blockDim.x + threadIdx.x;
    if (id >= N * 128) return;
    int n = id >> 7, d = id & 127;
    float s = dinv[n];
    agg[id] = g[id] * (s * s) + b_gcn[d];
}

// One wave per edge: agg[dst,:] += g[src,:] * dinv[src]*dinv[dst]
__global__ void k_agg_edges(const int* __restrict__ ei, const float* __restrict__ g,
                            const float* __restrict__ dinv, float* __restrict__ agg, int E) {
    int we = blockIdx.x * 8 + (threadIdx.x >> 5);
    if (we >= E) return;
    int lane = threadIdx.x & 31;
    int s = ei[we];
    int d = ei[(size_t)E + we];
    float w = dinv[s] * dinv[d];
    v4f v = ((const v4f*)(g + (size_t)s * 128))[lane];
    float* dst = agg + (size_t)d * 128 + lane * 4;
    atomicAdd(dst + 0, v[0] * w);
    atomicAdd(dst + 1, v[1] * w);
    atomicAdd(dst + 2, v[2] * w);
    atomicAdd(dst + 3, v[3] * w);
}

__global__ void k_reparam(const float* __restrict__ mu, const float* __restrict__ sd,
                          const float* __restrict__ eps, float* __restrict__ h, int total) {
    int id = blockIdx.x * blockDim.x + threadIdx.x;
    if (id >= total) return;
    h[id] = mu[id] + sd[id] * eps[id];
}

__global__ void k_pool(const float* __restrict__ yp, const int* __restrict__ batch,
                       float* __restrict__ sums, int N) {
    int id = blockIdx.x * blockDim.x + threadIdx.x;
    if (id >= N * 10) return;
    int n = id / 10, d = id - n * 10;
    atomicAdd(&sums[batch[n] * 10 + d], yp[id]);
}

__global__ void k_final(const float* __restrict__ sums, const float* __restrict__ cnt,
                        const float* __restrict__ y, float* __restrict__ out_ypred,
                        float* __restrict__ out_y, int G) {
    int id = blockIdx.x * blockDim.x + threadIdx.x;
    if (id >= G * 10) return;
    int g = id / 10;
    out_ypred[id] = sums[id] / fmaxf(cnt[g], 1.0f);
    out_y[id]     = y[id];
}

// ---------------------------------------------------------------------------
// Host orchestration
// ---------------------------------------------------------------------------
extern "C" void kernel_launch(void* const* d_in, const int* in_sizes, int n_in,
                              void* d_out, int out_size, void* d_ws, size_t ws_size,
                              hipStream_t stream) {
    const float* x     = (const float*)d_in[0];
    const int*   ei    = (const int*)  d_in[1];
    const int*   batch = (const int*)  d_in[2];
    const float* y     = (const float*)d_in[3];
    const float* eps   = (const float*)d_in[4];
    const float* W_in  = (const float*)d_in[5];  const float* b_in  = (const float*)d_in[6];
    const float* W_gcn = (const float*)d_in[7];  const float* b_gcn = (const float*)d_in[8];
    const float* W_enc = (const float*)d_in[9];  const float* b_enc = (const float*)d_in[10];
    const float* W_mu  = (const float*)d_in[11]; const float* b_mu  = (const float*)d_in[12];
    const float* W_std = (const float*)d_in[13]; const float* b_std = (const float*)d_in[14];
    const float* Wd0 = (const float*)d_in[15]; const float* bd0 = (const float*)d_in[16];
    const float* Wd1 = (const float*)d_in[17]; const float* bd1 = (const float*)d_in[18];
    const float* Wd2 = (const float*)d_in[19]; const float* bd2 = (const float*)d_in[20];
    const float* Wd3 = (const float*)d_in[21]; const float* bd3 = (const float*)d_in[22];
    const float* Wd4 = (const float*)d_in[23]; const float* bd4 = (const float*)d_in[24];
    const float* Wo  = (const float*)d_in[25]; const float* bo  = (const float*)d_in[26];

    const int N = in_sizes[0] / 32;
    const int E = in_sizes[1] / 2;
    const int G = in_sizes[3] / 10;
    const int K = in_sizes[4] / (N * 128);

    // ---- workspace bump allocator ----
    char*  ws  = (char*)d_ws;
    size_t off = 0;
    auto alloc = [&](size_t bytes) -> void* {
        void* p = ws + off;
        off += (bytes + 255) & ~(size_t)255;
        return p;
    };
    float* deg  = (float*)alloc((size_t)N * 4);
    float* dinv = (float*)alloc((size_t)N * 4);
    float* cnt  = (float*)alloc((size_t)G * 4);
    float* B0   = (float*)alloc((size_t)N * 128 * 4);
    float* B1   = (float*)alloc((size_t)N * 128 * 4);
    float* B2   = (float*)alloc((size_t)N * 128 * 4);
    float* B3   = (float*)alloc((size_t)N * 256 * 4);
    float* yp   = (float*)alloc((size_t)N * 10 * 4);
    float* sums = (float*)alloc((size_t)G * 10 * 4);
    // packed weights: bytes = KC*NT*32 fragments * 32B
    auto allocW = [&](int KC, int NT) -> v16bf* {
        return (v16bf*)alloc((size_t)KC * NT * 32 * 32);
    };
    v16bf* Wp_in  = allocW(1, 8);
    v16bf* Wp_gcn = allocW(4, 8);
    v16bf* Wp_enc = allocW(4, 8 * 5);   // 5 slices of 4x8
    v16bf* Wp_mu  = allocW(4, 8);
    v16bf* Wp_std = allocW(4, 8);
    v16bf* Wp_d0  = allocW(4, 16);
    v16bf* Wp_d1  = allocW(8, 8);
    v16bf* Wp_d2  = allocW(4, 4);
    v16bf* Wp_d3  = allocW(2, 2);
    v16bf* Wp_d4  = allocW(1, 1);
    v16bf* Wp_o   = allocW(1, 1);

    // ---- d_out layout: [ypred G*10 | mu N*128 | std N*128 | y G*10] ----
    float* out       = (float*)d_out;
    float* out_ypred = out;
    float* mu_out    = out + (size_t)G * 10;
    float* std_out   = mu_out + (size_t)N * 128;
    float* out_y     = std_out + (size_t)N * 128;

    hipMemsetAsync(deg,  0, (size_t)N * 4,      stream);
    hipMemsetAsync(cnt,  0, (size_t)G * 4,      stream);
    hipMemsetAsync(sums, 0, (size_t)G * 10 * 4, stream);

    // ---- pack weights into WMMA B-fragment layout (bf16, zero-padded) ----
    auto pack = [&](const float* W, v16bf* Wp, int Kr, int Nr, int KC, int NT) {
        int total = KC * NT * 32;
        k_pack<<<(total + 255) / 256, 256, 0, stream>>>(W, Wp, Kr, Nr, KC, NT);
    };
    pack(W_in,  Wp_in,  32, 128, 1, 8);
    pack(W_gcn, Wp_gcn, 128, 128, 4, 8);
    for (int j = 0; j < 5; ++j)
        pack(W_enc + (size_t)j * 128 * 128, Wp_enc + (size_t)j * 4 * 8 * 32, 128, 128, 4, 8);
    pack(W_mu,  Wp_mu,  128, 128, 4, 8);
    pack(W_std, Wp_std, 128, 128, 4, 8);
    pack(Wd0, Wp_d0, 128, 256, 4, 16);
    pack(Wd1, Wp_d1, 256, 128, 8, 8);
    pack(Wd2, Wp_d2, 128, 64, 4, 4);
    pack(Wd3, Wp_d3, 64, 32, 2, 2);
    pack(Wd4, Wp_d4, 32, 16, 1, 1);
    pack(Wo,  Wp_o,  16, 10, 1, 1);   // K=16->32 and N=10->16 zero-padded

    // ---- GCN normalization terms ----
    k_degree<<<(E + 255) / 256, 256, 0, stream>>>(ei, deg, E);
    k_dinv<<<(N + 255) / 256, 256, 0, stream>>>(deg, dinv, batch, cnt, N);

    const int gb  = (N + 127) / 128;          // GEMM row blocks
    const int geN = (N * 128 + 255) / 256;    // elementwise over [N,128]

    // ---- input layer: h = sigmoid(x @ W_in + b_in) ----
    k_gemm<1, 8, ACT_SIG, true, false><<<gb, 256, 0, stream>>>(x, 32, Wp_in, b_in, B0, 128, N, 128);

    // ---- K message-passing hops ----
    for (int i = 0; i < K; ++i) {
        k_gemm<4, 8, ACT_NONE, false, false><<<gb, 256, 0, stream>>>(B0, 128, Wp_gcn, nullptr, B1, 128, N, 128);
        k_agg_init<<<geN, 256, 0, stream>>>(B1, dinv, b_gcn, B2, N);
        k_agg_edges<<<(E + 7) / 8, 256, 0, stream>>>(ei, B1, dinv, B2, E);
        float* s = B2; float* d = B1;
        for (int j = 0; j < 5; ++j) {
            k_gemm<4, 8, ACT_SIG, true, false><<<gb, 256, 0, stream>>>(
                s, 128, Wp_enc + (size_t)j * 4 * 8 * 32, b_enc + (size_t)j * 128, d, 128, N, 128);
            float* t = s; s = d; d = t;
        }
        // after 5 layers result lives in s == B1
        k_gemm<4, 8, ACT_NONE, true, false><<<gb, 256, 0, stream>>>(s, 128, Wp_mu,  b_mu,  mu_out,  128, N, 128);
        k_gemm<4, 8, ACT_SP5,  true, false><<<gb, 256, 0, stream>>>(s, 128, Wp_std, b_std, std_out, 128, N, 128);
        k_reparam<<<geN, 256, 0, stream>>>(mu_out, std_out, eps + (size_t)i * N * 128, B0, N * 128);
    }

    // ---- decoder MLP ----
    k_gemm<4, 16, ACT_RELU, true, false><<<gb, 256, 0, stream>>>(B0, 128, Wp_d0, bd0, B3, 256, N, 256);
    k_gemm<8, 8,  ACT_RELU, true, false><<<gb, 256, 0, stream>>>(B3, 256, Wp_d1, bd1, B1, 128, N, 128);
    k_gemm<4, 4,  ACT_RELU, true, false><<<gb, 256, 0, stream>>>(B1, 128, Wp_d2, bd2, B2, 64,  N, 64);
    k_gemm<2, 2,  ACT_RELU, true, false><<<gb, 256, 0, stream>>>(B2, 64,  Wp_d3, bd3, B3, 32,  N, 32);
    hipMemsetAsync(B1, 0, (size_t)N * 128 * 4, stream);  // clean pad cols for the K=16->32 layer
    k_gemm<1, 1,  ACT_RELU, true, false><<<gb, 256, 0, stream>>>(B3, 32, Wp_d4, bd4, B1, 32, N, 16);
    k_gemm<1, 1,  ACT_SIG,  true, true ><<<gb, 256, 0, stream>>>(B1, 32, Wp_o,  bo,  yp, 10, N, 10);

    // ---- global mean pool + output assembly ----
    k_pool<<<(N * 10 + 255) / 256, 256, 0, stream>>>(yp, batch, sums, N);
    k_final<<<(G * 10 + 255) / 256, 256, 0, stream>>>(sums, cnt, y, out_ypred, out_y, G);

    (void)n_in; (void)out_size; (void)ws_size;
}
